// Transformers_43696997269572
// MI455X (gfx1250) — compile-verified, matrix operand order checked by reading, not run on hardware
//
#include <hip/hip_runtime.h>

typedef _Float16 f16;
typedef _Float16 v16h __attribute__((ext_vector_type(16)));
typedef float    v8f  __attribute__((ext_vector_type(8)));
typedef int      i32x4 __attribute__((ext_vector_type(4)));

// ---------------- problem constants ----------------
constexpr int NLAYERS = 2;
constexpr int HID     = 512;
constexpr int FFND    = 2048;
constexpr int NHEADS  = 8;
constexpr int SLEN    = 2048;          // 32*64
constexpr int BSZ     = 2;
constexpr int KD      = 64;            // HID/HEADS
constexpr int VD      = 1024;          // HID*2
constexpr int HS      = 128;           // VD/HEADS
constexpr int MROWS   = BSZ * SLEN;    // 4096
constexpr float SCALING = 0.125f;      // KD^-0.5
constexpr float LN_EPS  = 1e-5f;

// ---------------- GEMM tiling ----------------
constexpr int BM = 128, BN = 64, BK = 32;
constexpr int LDT = BK + 8;            // padded LDS stride (f16 elems), keeps 16B align

// CDNA5 async global->LDS copy (tracked by ASYNCcnt), with safe fallback.
#if __has_builtin(__builtin_amdgcn_global_load_async_to_lds_b128)
#define USE_ASYNC_LDS 1
#else
#define USE_ASYNC_LDS 0
#endif

#if USE_ASYNC_LDS
typedef __attribute__((address_space(1))) i32x4* as1_v4i;
typedef __attribute__((address_space(3))) i32x4* as3_v4i;
#endif

__device__ __forceinline__ void copy16_to_lds(f16* lds_dst, const f16* gsrc)
{
#if USE_ASYNC_LDS
    __builtin_amdgcn_global_load_async_to_lds_b128(
        (as1_v4i)(void*)gsrc,
        (as3_v4i)(void*)lds_dst,
        0, 0);
#else
    *reinterpret_cast<uint4*>(lds_dst) = *reinterpret_cast<const uint4*>(gsrc);
#endif
}

__device__ __forceinline__ void wait_async_copies()
{
#if USE_ASYNC_LDS
 #if __has_builtin(__builtin_amdgcn_s_wait_asynccnt)
    __builtin_amdgcn_s_wait_asynccnt(0);
 #else
    asm volatile("s_wait_asynccnt 0" ::: "memory");
 #endif
#endif
}

// C(MxN) = A(MxK) * B^T, A row-major f16, B given as N x K row-major f16.
// Epilogue: optional bias (f32, indexed by col), ReLU, residual add (f32, C layout),
// stores f32 and/or f16.
__global__ __launch_bounds__(256)
void gemm_wmma(const f16* __restrict__ A, const f16* __restrict__ B,
               int M, int N, int K, int lda, int ldb, int ldc,
               const float* __restrict__ bias,
               const float* __restrict__ residual,
               float* __restrict__ C32, f16* __restrict__ C16,
               int relu)
{
    __shared__ f16 As[2][BM * LDT];
    __shared__ f16 Bs[2][BN * LDT];

    const int tid  = threadIdx.x;
    const int lane = tid & 31;
    const int wid  = tid >> 5;
    const int wm   = wid & 3;          // 4 waves along M
    const int wn   = wid >> 2;         // 2 waves along N
    const int m0   = blockIdx.y * BM;
    const int n0   = blockIdx.x * BN;

    const f16* Ag = A + (size_t)m0 * lda;
    const f16* Bg = B + (size_t)n0 * ldb;

    auto stage = [&](int k0, int buf) {
        // A tile: 128 rows x 32 f16 = 512 x 16B chunks -> 2 per thread
        #pragma unroll
        for (int it = 0; it < 2; ++it) {
            int idx = tid + it * 256;
            int r = idx >> 2, c = (idx & 3) * 8;
            copy16_to_lds(&As[buf][r * LDT + c], Ag + (size_t)r * lda + k0 + c);
        }
        // B tile: 64 rows x 32 f16 = 256 x 16B chunks -> 1 per thread
        int r = tid >> 2, c = (tid & 3) * 8;
        copy16_to_lds(&Bs[buf][r * LDT + c], Bg + (size_t)r * ldb + k0 + c);
    };

    union Frag { v16h v; uint4 q[2]; };
    v8f acc[2][2] = {};

    const int koff = (lane >> 4) * 8;  // K sub-block select per ISA 16-bit layout
    const int rsel = lane & 15;        // row (A) / col (B) within 16

    stage(0, 0);
    wait_async_copies();
    __syncthreads();
    int buf = 0;

    for (int kk = 0; kk < K; kk += BK) {
        if (kk + BK < K) {
            stage(kk + BK, buf ^ 1);           // async copy overlaps WMMA below
            if (kk + 2 * BK < K)
                __builtin_prefetch(Ag + (size_t)(tid >> 1) * lda + kk + 2 * BK, 0, 1);
        }
        Frag a[2], b[2];
        #pragma unroll
        for (int mi = 0; mi < 2; ++mi) {
            const f16* p = &As[buf][(wm * 32 + mi * 16 + rsel) * LDT + koff];
            a[mi].q[0] = *reinterpret_cast<const uint4*>(p);
            a[mi].q[1] = *reinterpret_cast<const uint4*>(p + 16);
        }
        #pragma unroll
        for (int ni = 0; ni < 2; ++ni) {
            const f16* p = &Bs[buf][(wn * 32 + ni * 16 + rsel) * LDT + koff];
            b[ni].q[0] = *reinterpret_cast<const uint4*>(p);
            b[ni].q[1] = *reinterpret_cast<const uint4*>(p + 16);
        }
        #pragma unroll
        for (int mi = 0; mi < 2; ++mi)
            #pragma unroll
            for (int ni = 0; ni < 2; ++ni)
                acc[mi][ni] = __builtin_amdgcn_wmma_f32_16x16x32_f16(
                    false, a[mi].v, false, b[ni].v,
                    (short)0, acc[mi][ni], false, false);
        wait_async_copies();
        __syncthreads();
        buf ^= 1;
    }

    // epilogue: C VGPR layout -> M = r + 8*(lane>=16), N = lane&15
    #pragma unroll
    for (int mi = 0; mi < 2; ++mi) {
        #pragma unroll
        for (int ni = 0; ni < 2; ++ni) {
            int col = n0 + wn * 32 + ni * 16 + (lane & 15);
            float bv = bias ? bias[col] : 0.0f;
            #pragma unroll
            for (int r = 0; r < 8; ++r) {
                int row = m0 + wm * 32 + mi * 16 + r + ((lane >> 4) << 3);
                float v = acc[mi][ni][r] + bv;
                if (relu) v = fmaxf(v, 0.0f);
                size_t cidx = (size_t)row * ldc + col;
                if (residual) v += residual[cidx];
                if (C32) C32[cidx] = v;
                if (C16) C16[cidx] = (f16)v;
            }
        }
    }
}

// ---------------- elementwise / reduction kernels ----------------

__global__ void cvt_copy_kernel(const float* __restrict__ in,
                                float* __restrict__ o32, f16* __restrict__ o16, int n)
{
    int i = blockIdx.x * blockDim.x + threadIdx.x;
    if (i < n) { float v = in[i]; if (o32) o32[i] = v; o16[i] = (f16)v; }
}

// W (K x N) f32 row-major  ->  Wt (N x K) f16 row-major
__global__ void wcvt_t_kernel(const float* __restrict__ W, f16* __restrict__ Wt, int K, int N)
{
    int i = blockIdx.x * blockDim.x + threadIdx.x;
    if (i < K * N) {
        int k = i / N, n = i % N;
        Wt[(size_t)n * K + k] = (f16)W[i];
    }
}

// V (B, S, VD) f16  ->  Vt (B, HEADS, HS, S) f16  (per-head transpose)
__global__ void vtrans_kernel(const f16* __restrict__ V, f16* __restrict__ Vt)
{
    int idx = blockIdx.x * blockDim.x + threadIdx.x;   // B*S*VD = 2^22
    int n = idx & (VD - 1);
    int s = (idx >> 10) & (SLEN - 1);
    int b = idx >> 21;
    int h = n >> 7, c = n & (HS - 1);
    Vt[((size_t)((b * NHEADS + h) * HS + c)) * SLEN + s] = V[idx];
}

// xPos rotary applied in place to (B, S, HEADS, KD) f16
__global__ void rotary_kernel(f16* __restrict__ Q)
{
    int idx = blockIdx.x * blockDim.x + threadIdx.x;   // 2^20 pairs
    int i = idx & 31;
    int h = (idx >> 5) & 7;
    int s = (idx >> 8) & 2047;
    int b = idx >> 19;
    size_t base = ((size_t)(b * SLEN + s)) * HID + h * KD + 2 * i;
    float bse   = (2.0f * i + 0.4f * KD) / (1.4f * KD);
    float pos   = (float)(s - SLEN / 2);
    float scale = powf(bse, pos * (1.0f / 512.0f));
    float invf  = __expf(-((float)i / 32.0f) * logf(10000.0f));
    float th    = (float)s * invf;
    float sn = __sinf(th) * scale, cs = __cosf(th) * scale;
    float x1 = (float)Q[base], x2 = (float)Q[base + 1];
    Q[base]     = (f16)(x1 * cs - x2 * sn);
    Q[base + 1] = (f16)(x2 * cs + x1 * sn);
}

// softmax over rows of length 2048 (in-place f16), folds SCALING
__global__ __launch_bounds__(256) void softmax_kernel(f16* __restrict__ P)
{
    __shared__ float red[8];
    size_t base = (size_t)blockIdx.x * 2048;
    int tid = threadIdx.x, lane = tid & 31, wid = tid >> 5;
    float v[8];
    #pragma unroll
    for (int j = 0; j < 8; ++j) v[j] = (float)P[base + tid + j * 256];
    float m = v[0];
    #pragma unroll
    for (int j = 1; j < 8; ++j) m = fmaxf(m, v[j]);
    for (int o = 16; o > 0; o >>= 1) m = fmaxf(m, __shfl_xor(m, o, 32));
    if (lane == 0) red[wid] = m;
    __syncthreads();
    float mm = red[0];
    #pragma unroll
    for (int w = 1; w < 8; ++w) mm = fmaxf(mm, red[w]);
    __syncthreads();
    float s = 0.0f;
    #pragma unroll
    for (int j = 0; j < 8; ++j) { v[j] = __expf(v[j] - mm); s += v[j]; }
    for (int o = 16; o > 0; o >>= 1) s += __shfl_xor(s, o, 32);
    if (lane == 0) red[wid] = s;
    __syncthreads();
    float ss = 0.0f;
    #pragma unroll
    for (int w = 0; w < 8; ++w) ss += red[w];
    float inv = SCALING / ss;
    #pragma unroll
    for (int j = 0; j < 8; ++j) P[base + tid + j * 256] = (f16)(v[j] * inv);
}

// LayerNorm over rows of 512; writes f32 (+optional f16 copy)
__global__ __launch_bounds__(256)
void layernorm_kernel(const float* __restrict__ Z, const float* __restrict__ g,
                      const float* __restrict__ be,
                      float* __restrict__ o32, f16* __restrict__ o16)
{
    __shared__ float rs[8], rq[8];
    size_t base = (size_t)blockIdx.x * HID;
    int tid = threadIdx.x, lane = tid & 31, wid = tid >> 5;
    float a = Z[base + tid], b = Z[base + tid + 256];
    float s = a + b, q = a * a + b * b;
    for (int o = 16; o > 0; o >>= 1) {
        s += __shfl_xor(s, o, 32);
        q += __shfl_xor(q, o, 32);
    }
    if (lane == 0) { rs[wid] = s; rq[wid] = q; }
    __syncthreads();
    float S = 0.0f, Q = 0.0f;
    #pragma unroll
    for (int w = 0; w < 8; ++w) { S += rs[w]; Q += rq[w]; }
    float mean = S * (1.0f / HID);
    float var  = Q * (1.0f / HID) - mean * mean;
    float inv  = rsqrtf(var + LN_EPS);
    float y0 = (a - mean) * inv * g[tid] + be[tid];
    float y1 = (b - mean) * inv * g[tid + 256] + be[tid + 256];
    o32[base + tid]       = y0;
    o32[base + tid + 256] = y1;
    if (o16) {
        o16[base + tid]       = (f16)y0;
        o16[base + tid + 256] = (f16)y1;
    }
}

// ---------------- host orchestration ----------------

static void launch_gemm(hipStream_t st, const f16* A, const f16* B,
                        int M, int N, int K, int lda, int ldb, int ldc,
                        const float* bias, const float* res,
                        float* C32, f16* C16, int relu)
{
    dim3 grid(N / BN, M / BM, 1);
    gemm_wmma<<<grid, 256, 0, st>>>(A, B, M, N, K, lda, ldb, ldc,
                                    bias, res, C32, C16, relu);
}

static inline size_t al256(size_t x) { return (x + 255) & ~(size_t)255; }

extern "C" void kernel_launch(void* const* d_in, const int* in_sizes, int n_in,
                              void* d_out, int out_size, void* d_ws, size_t ws_size,
                              hipStream_t stream)
{
    const float* x_in = (const float*)d_in[0];
    const float* Wq   = (const float*)d_in[1];
    const float* Wk   = (const float*)d_in[2];
    const float* Wv   = (const float*)d_in[3];
    const float* Wo   = (const float*)d_in[4];
    const float* W1   = (const float*)d_in[5];
    const float* b1   = (const float*)d_in[6];
    const float* W2   = (const float*)d_in[7];
    const float* b2   = (const float*)d_in[8];
    const float* g1   = (const float*)d_in[9];
    const float* be1  = (const float*)d_in[10];
    const float* g2   = (const float*)d_in[11];
    const float* be2  = (const float*)d_in[12];
    float* out = (float*)d_out;

    char* base = (char*)d_ws;
    size_t off = 0;
    auto alloc = [&](size_t bytes) { void* r = base + off; off = al256(off + bytes); return r; };

    float* X32  = (float*)alloc((size_t)MROWS * HID * 4);
    float* Z32  = (float*)alloc((size_t)MROWS * HID * 4);
    float* Yn32 = (float*)alloc((size_t)MROWS * HID * 4);
    f16* X16   = (f16*)alloc((size_t)MROWS * HID * 2);
    f16* Yn16  = (f16*)alloc((size_t)MROWS * HID * 2);
    f16* Q16   = (f16*)alloc((size_t)MROWS * HID * 2);
    f16* K16   = (f16*)alloc((size_t)MROWS * HID * 2);
    f16* V16   = (f16*)alloc((size_t)MROWS * VD * 2);
    f16* Vt16  = (f16*)alloc((size_t)MROWS * VD * 2);
    f16* Att16 = (f16*)alloc((size_t)MROWS * VD * 2);
    f16* H1    = (f16*)alloc((size_t)MROWS * FFND * 2);
    f16* Pbuf  = (f16*)alloc((size_t)SLEN * SLEN * 2);        // one (b,h) score tile
    f16* Wq_t  = (f16*)alloc((size_t)HID * HID * 2);
    f16* Wk_t  = (f16*)alloc((size_t)HID * HID * 2);
    f16* Wv_t  = (f16*)alloc((size_t)HID * VD * 2);
    f16* Wo_t  = (f16*)alloc((size_t)VD * HID * 2);
    f16* W1_t  = (f16*)alloc((size_t)HID * FFND * 2);
    f16* W2_t  = (f16*)alloc((size_t)FFND * HID * 2);

    cvt_copy_kernel<<<(MROWS * HID) / 256, 256, 0, stream>>>(x_in, X32, X16, MROWS * HID);

    for (int i = 0; i < NLAYERS; ++i) {
        wcvt_t_kernel<<<(HID * HID)  / 256, 256, 0, stream>>>(Wq + (size_t)i * HID * HID,  Wq_t, HID,  HID);
        wcvt_t_kernel<<<(HID * HID)  / 256, 256, 0, stream>>>(Wk + (size_t)i * HID * HID,  Wk_t, HID,  HID);
        wcvt_t_kernel<<<(HID * VD)   / 256, 256, 0, stream>>>(Wv + (size_t)i * HID * VD,   Wv_t, HID,  VD);
        wcvt_t_kernel<<<(VD * HID)   / 256, 256, 0, stream>>>(Wo + (size_t)i * VD * HID,   Wo_t, VD,   HID);
        wcvt_t_kernel<<<(HID * FFND) / 256, 256, 0, stream>>>(W1 + (size_t)i * HID * FFND, W1_t, HID,  FFND);
        wcvt_t_kernel<<<(FFND * HID) / 256, 256, 0, stream>>>(W2 + (size_t)i * FFND * HID, W2_t, FFND, HID);

        // Q/K/V projections
        launch_gemm(stream, X16, Wq_t, MROWS, HID, HID, HID, HID, HID,
                    nullptr, nullptr, nullptr, Q16, 0);
        launch_gemm(stream, X16, Wk_t, MROWS, HID, HID, HID, HID, HID,
                    nullptr, nullptr, nullptr, K16, 0);
        launch_gemm(stream, X16, Wv_t, MROWS, VD, HID, HID, HID, VD,
                    nullptr, nullptr, nullptr, V16, 0);

        rotary_kernel<<<(BSZ * SLEN * NHEADS * 32) / 256, 256, 0, stream>>>(Q16);
        rotary_kernel<<<(BSZ * SLEN * NHEADS * 32) / 256, 256, 0, stream>>>(K16);
        vtrans_kernel<<<(BSZ * SLEN * VD) / 256, 256, 0, stream>>>(V16, Vt16);

        // attention per (b, h): scores -> softmax -> att @ V
        for (int b = 0; b < BSZ; ++b) {
            for (int h = 0; h < NHEADS; ++h) {
                const f16* Qh  = Q16  + (size_t)b * SLEN * HID + h * KD;
                const f16* Kh  = K16  + (size_t)b * SLEN * HID + h * KD;
                const f16* Vth = Vt16 + (size_t)(b * NHEADS + h) * HS * SLEN;
                f16* Oh = Att16 + (size_t)b * SLEN * VD + (size_t)h * HS;
                // scores: Q (S x KD) @ K^T ; K row-major == N x K layout
                launch_gemm(stream, Qh, Kh, SLEN, SLEN, KD, HID, HID, SLEN,
                            nullptr, nullptr, nullptr, Pbuf, 0);
                softmax_kernel<<<SLEN, 256, 0, stream>>>(Pbuf);
                // att @ V ; Vt is N x K layout (HS x S)
                launch_gemm(stream, Pbuf, Vth, SLEN, HS, SLEN, SLEN, SLEN, VD,
                            nullptr, nullptr, nullptr, Oh, 0);
            }
        }

        // output projection + residual
        launch_gemm(stream, Att16, Wo_t, MROWS, HID, VD, VD, VD, HID,
                    nullptr, X32, Z32, nullptr, 0);
        layernorm_kernel<<<MROWS, 256, 0, stream>>>(Z32, g1 + (size_t)i * HID, be1 + (size_t)i * HID,
                                                    Yn32, Yn16);

        // FFN
        launch_gemm(stream, Yn16, W1_t, MROWS, FFND, HID, HID, HID, FFND,
                    b1 + (size_t)i * FFND, nullptr, nullptr, H1, 1);
        launch_gemm(stream, H1, W2_t, MROWS, HID, FFND, FFND, FFND, HID,
                    b2 + (size_t)i * HID, Yn32, Z32, nullptr, 0);

        float* lnOut = (i == NLAYERS - 1) ? out : X32;
        layernorm_kernel<<<MROWS, 256, 0, stream>>>(Z32, g2 + (size_t)i * HID, be2 + (size_t)i * HID,
                                                    lnOut, X16);
    }
}
